// VQ_VAELoss_70274254897675
// MI455X (gfx1250) — compile-verified, matrix operand order checked by reading, not run on hardware
//
#include <hip/hip_runtime.h>
#include <hip/hip_bf16.h>

// ---------------------------------------------------------------------------
// VQ-VAE loss on gfx1250 (MI455X).
// Problem sizes (fixed by the reference): N=32, d=64, H=W=32, K=2048.
// tokens = N*H*W = 32768.  Distance GEMM: 32768 x 2048 x 64.
// Strategy: bf16 hi/lo split GEMM on v_wmma_f32_16x16x32_bf16 (XDL pipe),
// M-blocked 2x per wave (24 FLOP/B -> 47 FLOP/B from L2), fused row-argmin,
// then a light gather+reduction pass.
// ---------------------------------------------------------------------------

typedef __attribute__((ext_vector_type(16))) __bf16 v16bf;
typedef __attribute__((ext_vector_type(8)))  float  v8f;

static constexpr int kNB     = 32;    // batch
static constexpr int kD      = 64;    // channels
static constexpr int kSP     = 1024;  // H*W
static constexpr int kK      = 2048;  // codebook size
static constexpr int kTokens = kNB * kSP;       // 32768
static constexpr int kChunks = kK / 16;         // 128 code chunks of 16
static constexpr int kMT     = 2;               // M tiles (16 tokens each) per wave

__device__ __forceinline__ v8f wmma_bf16(v16bf a, v16bf b, v8f c) {
  // v_wmma_f32_16x16x32_bf16  (8-arg form, probe-confirmed)
  return __builtin_amdgcn_wmma_f32_16x16x32_bf16(
      /*neg_a=*/false, a, /*neg_b=*/false, b,
      /*c_mod=*/(short)0, c, /*reuse_a=*/false, /*reuse_b=*/false);
}

// ---------------------------------------------------------------------------
// Kernel 0: transform codebook (K x d, f32) into pre-swizzled WMMA B-fragments
// (bf16 hi + bf16 lo) laid out exactly as v_wmma_f32_16x16x32_bf16 wants its
// B operand per lane, so the hot loop loads fragments with coalesced B128s.
//
// Fragment id = (chunk*2 + kstep)*2 + h   (h: 0=hi, 1=lo), 512 bf16 each.
// 16-bit operand layout (wave32): lanes 0-15 hold column n (=lane), slots
// j<8 -> K=j, j>=8 -> K=j+8 ; lanes 16-31 hold n=lane-16, K shifted by +8.
// ---------------------------------------------------------------------------
__global__ void vq_prep_frags(const float* __restrict__ codebook,
                              __bf16* __restrict__ frags) {
  int gid = blockIdx.x * blockDim.x + threadIdx.x;   // 0 .. K*d-1
  if (gid >= kK * kD) return;
  int n = gid >> 6;          // code index
  int k = gid & 63;          // channel
  float e = codebook[gid];
  __bf16 eh = (__bf16)e;
  __bf16 el = (__bf16)(e - (float)eh);

  int c  = n >> 4;           // chunk
  int nl = n & 15;           // column within chunk
  int s  = k >> 5;           // kstep (0: K 0-31, 1: K 32-63)
  int kk = k & 31;

  int laneHalf, j;
  if      (kk <  8) { laneHalf = 0; j = kk;      }
  else if (kk < 16) { laneHalf = 1; j = kk - 8;  }
  else if (kk < 24) { laneHalf = 0; j = kk - 8;  }
  else              { laneHalf = 1; j = kk - 16; }
  int lane = nl + 16 * laneHalf;

  size_t base = ((size_t)(c * 2 + s) * 2) * 512 + (size_t)lane * 16 + j;
  frags[base]       = eh;   // h = 0
  frags[base + 512] = el;   // h = 1
}

// Kernel 0b: per-code squared norms (argmin only needs ||e||^2 - 2 x.e).
__global__ void vq_code_norms(const float* __restrict__ codebook,
                              float* __restrict__ cbNorm) {
  int n = blockIdx.x * blockDim.x + threadIdx.x;
  if (n >= kK) return;
  const float* row = codebook + (size_t)n * kD;
  float s = 0.f;
#pragma unroll
  for (int k = 0; k < kD; ++k) { float v = row[k]; s += v * v; }
  cbNorm[n] = s;
}

// ---------------------------------------------------------------------------
// Kernel A: one wave per kMT*16-token group. A fragments (hi/lo) for both
// M tiles cached in VGPRs; 128 chunks x (6*kMT) WMMAs, each B fragment
// reused kMT times; fused running argmin per tile.
// Block = 256 threads = 8 waves; grid = 128 -> 1024 groups = 32768 tokens.
// ---------------------------------------------------------------------------
__global__ void __launch_bounds__(256)
vq_argmin(const float* __restrict__ latents,
          const float* __restrict__ cbNorm,
          const __bf16* __restrict__ frags,
          int* __restrict__ outIdx) {
  const int tid       = threadIdx.x;
  const int lane      = tid & 31;
  const int wave      = tid >> 5;
  const int group     = blockIdx.x * 8 + wave;       // 0..1023
  const int tokenBase = group * (kMT * 16);

  const int laneHalf = lane >> 4;                    // 0/1
  const int mrow     = lane & 15;                    // matrix row owned (A side)
  const int col      = lane & 15;                    // C/D column owned

  // ---- load A fragments (kMT x 16 tokens x 64 ch) as bf16 hi/lo
  v16bf ah[kMT][2], al[kMT][2];
#pragma unroll
  for (int t = 0; t < kMT; ++t) {
    // token = n*1024 + spatial; a 16-token tile never crosses n (16 | 1024)
    const int tb = tokenBase + t * 16;
    const int n  = tb >> 10;
    const int sB = tb & 1023;
    const float* xbase = latents + (size_t)n * (kD * kSP) + sB + mrow;
#pragma unroll
    for (int s2 = 0; s2 < 2; ++s2) {
#pragma unroll
      for (int j = 0; j < 16; ++j) {
        int kk = ((j < 8) ? j : (j + 8)) + laneHalf * 8;  // K within kstep
        float xv = xbase[(size_t)(s2 * 32 + kk) * kSP];
        __bf16 h = (__bf16)xv;
        ah[t][s2][j] = h;
        al[t][s2][j] = (__bf16)(xv - (float)h);
      }
    }
  }

  float best[kMT][8];
  int   bidx[kMT][8];
#pragma unroll
  for (int t = 0; t < kMT; ++t)
#pragma unroll
    for (int r = 0; r < 8; ++r) { best[t][r] = 3.4e38f; bidx[t][r] = 0; }

  for (int c = 0; c < kChunks; ++c) {
    const __bf16* fb = frags + (size_t)c * 2048 + (size_t)lane * 16;
    v16bf bh0 = *(const v16bf*)(fb);             // kstep 0, hi
    v16bf bl0 = *(const v16bf*)(fb + 512);       // kstep 0, lo
    v16bf bh1 = *(const v16bf*)(fb + 1024);      // kstep 1, hi
    v16bf bl1 = *(const v16bf*)(fb + 1536);      // kstep 1, lo
    __builtin_prefetch(fb + 2048, 0, 0);         // next chunk -> global_prefetch_b8

    const int   code = c * 16 + col;
    const float nrm  = cbNorm[code];

#pragma unroll
    for (int t = 0; t < kMT; ++t) {
      v8f acc = {};
      // x.e = xh*eh + xh*el + xl*eh  (xl*el term ~2^-16, dropped)
      acc = wmma_bf16(ah[t][0], bh0, acc);
      acc = wmma_bf16(ah[t][0], bl0, acc);
      acc = wmma_bf16(al[t][0], bh0, acc);
      acc = wmma_bf16(ah[t][1], bh1, acc);
      acc = wmma_bf16(ah[t][1], bl1, acc);
      acc = wmma_bf16(al[t][1], bh1, acc);
#pragma unroll
      for (int r = 0; r < 8; ++r) {
        float dv = nrm - 2.0f * acc[r];          // ||e||^2 - 2 x.e
        if (dv < best[t][r]) { best[t][r] = dv; bidx[t][r] = code; }
      }
    }
  }

  // reduce over the 16 lanes sharing each row (xor masks stay within half)
#pragma unroll
  for (int t = 0; t < kMT; ++t) {
#pragma unroll
    for (int off = 1; off < 16; off <<= 1) {
#pragma unroll
      for (int r = 0; r < 8; ++r) {
        float ob = __shfl_xor(best[t][r], off, 32);
        int   oi = __shfl_xor(bidx[t][r], off, 32);
        if (ob < best[t][r] || (ob == best[t][r] && oi < bidx[t][r])) {
          best[t][r] = ob; bidx[t][r] = oi;
        }
      }
    }
  }

  if (col == 0) {
#pragma unroll
    for (int t = 0; t < kMT; ++t)
#pragma unroll
      for (int r = 0; r < 8; ++r)    // rows: 0-7 (lane 0), 8-15 (lane 16)
        outIdx[tokenBase + t * 16 + r + 8 * laneHalf] = bidx[t][r];
  }
}

// ---------------------------------------------------------------------------
// Kernel Z: zero the two loss accumulators (must happen every call).
// ---------------------------------------------------------------------------
__global__ void vq_zero(float* __restrict__ accum) {
  if (threadIdx.x == 0) { accum[0] = 0.f; accum[1] = 0.f; }
}

// ---------------------------------------------------------------------------
// Kernel B: gather q = codebook[idx], fused:
//   S1 += ||q - x||^2            (codebook + commitment share this value)
//   y  = dec_w @ q + dec_b ; S2 += ||y - target||^2
// One thread per token; block reduction; one atomicAdd pair per block.
// ---------------------------------------------------------------------------
__global__ void __launch_bounds__(256)
vq_loss(const float* __restrict__ latents,
        const float* __restrict__ target,
        const float* __restrict__ codebook,
        const float* __restrict__ dec_w,
        const float* __restrict__ dec_b,
        const int* __restrict__ idx,
        float* __restrict__ accum) {
  const int t = blockIdx.x * blockDim.x + threadIdx.x;   // 0..32767
  const int n = t >> 10;
  const int s = t & 1023;
  const float* x = latents + (size_t)n * (kD * kSP) + s; // stride kSP per ch
  const float* q = codebook + (size_t)idx[t] * kD;

  float se = 0.f;
  float y0 = dec_b[0], y1 = dec_b[1], y2 = dec_b[2];
#pragma unroll 8
  for (int c = 0; c < kD; ++c) {
    float qc = q[c];
    float xc = x[(size_t)c * kSP];
    float dd = qc - xc;
    se += dd * dd;
    y0 += dec_w[c] * qc;
    y1 += dec_w[kD + c] * qc;
    y2 += dec_w[2 * kD + c] * qc;
  }
  const float* tg = target + (size_t)n * (3 * kSP) + s;
  float r0 = y0 - tg[0];
  float r1 = y1 - tg[kSP];
  float r2 = y2 - tg[2 * kSP];
  float re = r0 * r0 + r1 * r1 + r2 * r2;

  __shared__ float sSe[256];
  __shared__ float sRe[256];
  sSe[threadIdx.x] = se;
  sRe[threadIdx.x] = re;
  __syncthreads();
  for (int st = 128; st > 0; st >>= 1) {
    if (threadIdx.x < st) {
      sSe[threadIdx.x] += sSe[threadIdx.x + st];
      sRe[threadIdx.x] += sRe[threadIdx.x + st];
    }
    __syncthreads();
  }
  if (threadIdx.x == 0) {
    atomicAdd(&accum[0], sSe[0]);
    atomicAdd(&accum[1], sRe[0]);
  }
}

// Kernel C: finalize 3 scalars (total, vq_loss, recon).
__global__ void vq_finalize(const float* __restrict__ accum,
                            float* __restrict__ out) {
  if (threadIdx.x == 0) {
    float vq    = 2.0f * accum[0] / (float)(kNB * kD * kSP);   // codebook+commit
    float recon = accum[1] / (float)(kNB * 3 * kSP);
    out[0] = vq + recon;
    out[1] = vq;
    out[2] = recon;
  }
}

// ---------------------------------------------------------------------------
extern "C" void kernel_launch(void* const* d_in, const int* in_sizes, int n_in,
                              void* d_out, int out_size, void* d_ws, size_t ws_size,
                              hipStream_t stream) {
  (void)in_sizes; (void)n_in; (void)out_size; (void)ws_size;
  const float* latents  = (const float*)d_in[0];   // (32,64,32,32)
  const float* target   = (const float*)d_in[1];   // (32,3,32,32)
  const float* codebook = (const float*)d_in[2];   // (2048,64)
  const float* dec_w    = (const float*)d_in[3];   // (3,64)
  const float* dec_b    = (const float*)d_in[4];   // (3,)
  float* out = (float*)d_out;

  // workspace carve-out
  char* ws = (char*)d_ws;
  __bf16* frags  = (__bf16*)ws;                       // 512 frags * 512 bf16 = 512 KB
  float*  cbNorm = (float*)(ws + 512 * 1024);         //   8 KB
  int*    idxBuf = (int*)  (ws + 512 * 1024 + 8192);  // 128 KB
  float*  accum  = (float*)(ws + 512 * 1024 + 8192 + 128 * 1024); // 8 B

  vq_prep_frags<<<(kK * kD + 255) / 256, 256, 0, stream>>>(codebook, frags);
  vq_code_norms<<<(kK + 255) / 256, 256, 0, stream>>>(codebook, cbNorm);
  vq_zero<<<1, 64, 0, stream>>>(accum);
  vq_argmin<<<kTokens / (kMT * 16 * 8), 256, 0, stream>>>(latents, cbNorm, frags, idxBuf);
  vq_loss<<<kTokens / 256, 256, 0, stream>>>(latents, target, codebook,
                                             dec_w, dec_b, idxBuf, accum);
  vq_finalize<<<1, 64, 0, stream>>>(accum, out);
}
